// FP8FP4RadixAttention_40535901340411
// MI455X (gfx1250) — matrix operand order repaired and checked by reading, not compiled, not run on hardware
//
#include <hip/hip_runtime.h>

// ---------------- problem constants ----------------
constexpr int CB   = 2;
constexpr int CS   = 2048;
constexpr int CHQ  = 32;
constexpr int CHKV = 8;
constexpr int CGQ  = CHQ / CHKV;   // 4
constexpr int CD   = 128;
constexpr int QB   = 128;          // query rows per block (8 waves x 16)
constexpr int KT   = 32;           // keys per tile (WMMA K for PV)
constexpr int NWAVE = 8;
#define SCALING 0.08838834764831845f

typedef __attribute__((ext_vector_type(16))) _Float16 v16h;
typedef __attribute__((ext_vector_type(8)))  float    v8f;
typedef unsigned int tdm_v4u __attribute__((ext_vector_type(4)));
typedef int          tdm_v8i __attribute__((ext_vector_type(8)));
typedef int          tdm_v4i __attribute__((ext_vector_type(4)));

union AFrag { v16h v; _Float16 h[16]; };

// ---------------- dequant pre-pass ----------------
// e4m3 quant-dequant, then per-(token,head) absmax FP4 e2m1 quant-dequant.
// dstStride lets V be written transposed ([B][HKV][D][S]) for TDM-friendly tiles.
__device__ inline void qdq_row(const float* __restrict__ src,
                               _Float16* __restrict__ dst, long dstStride,
                               int lane) {
  float x[4];
  float amax = 0.f;
#pragma unroll
  for (int j = 0; j < 4; ++j) {
    float xx = src[lane * 4 + j];
    float ax = fabsf(xx);
    float xc = fminf(fmaxf(xx, -448.f), 448.f);
    float e  = fminf(fmaxf(floorf(log2f(fmaxf(ax, 1e-12f))), -6.f), 8.f);
    float q8 = rintf(xc * exp2f(3.f - e)) * exp2f(e - 3.f);
    q8 = (ax == 0.f) ? 0.f : q8;
    x[j] = q8;
    amax = fmaxf(amax, fabsf(q8));
  }
#pragma unroll
  for (int off = 1; off < 32; off <<= 1)
    amax = fmaxf(amax, __shfl_xor(amax, off, 32));
  float scale = fmaxf(amax * (1.0f / 6.0f), 1e-12f);
  float inv   = 1.0f / scale;
#pragma unroll
  for (int j = 0; j < 4; ++j) {
    float xn = x[j] * inv;
    float a  = fabsf(xn);
    float g = (a < 0.25f) ? 0.f  : (a < 0.75f) ? 0.5f : (a < 1.25f) ? 1.f  :
              (a < 1.75f) ? 1.5f : (a < 2.5f)  ? 2.f  : (a < 3.5f)  ? 3.f  :
              (a < 5.0f)  ? 4.f  : 6.f;
    dst[(long)(lane * 4 + j) * dstStride] = (_Float16)(copysignf(g, xn) * scale);
  }
}

__global__ __launch_bounds__(256)
void fp8fp4_dequant_kv(const float* __restrict__ k, const float* __restrict__ v,
                       _Float16* __restrict__ kd, _Float16* __restrict__ vdT,
                       int nrows) {
  int row  = blockIdx.x * (blockDim.x >> 5) + (threadIdx.x >> 5);
  int lane = threadIdx.x & 31;
  if (row >= nrows) return;
  // row = (b*S + s)*HKV + h
  int b = row / (CS * CHKV);
  int rem = row - b * CS * CHKV;
  int s = rem / CHKV;
  int h = rem - s * CHKV;
  qdq_row(k + (size_t)row * CD, kd + (size_t)row * CD, 1, lane);
  // V transposed: vdT[b][h][d][s]
  _Float16* vdst = vdT + ((size_t)(b * CHKV + h) * CD) * CS + s;
  qdq_row(v + (size_t)row * CD, vdst, CS, lane);
}

// ---------------- TDM: 2-D tile Global -> LDS (D# per ISA ch.8) ----------------
// dims/strides in elements of data_size=2B. tile rows contiguous (tile_d0),
// tile_d1 rows at stride0 elements.
__device__ inline void tdm_load_2d(unsigned lds_addr, const void* gptr,
                                   unsigned tile_d0, unsigned tile_d1,
                                   unsigned tensor_d0, unsigned tensor_d1,
                                   unsigned long long stride0) {
  unsigned long long ga = (unsigned long long)(uintptr_t)gptr;
  tdm_v4u g0;
  g0[0] = 1u;                                         // count=1, user descriptor
  g0[1] = lds_addr;                                   // LDS byte address
  g0[2] = (unsigned)(ga & 0xFFFFFFFFu);               // global_addr[31:0]
  g0[3] = (unsigned)((ga >> 32) & 0x01FFFFFFu)        // global_addr[56:32]
          | (2u << 30);                               // type = 2 ("image")
  tdm_v8i g1;
  g1[0] = (int)(1u << 16);                            // wg_mask=0, data_size=2B
  g1[1] = (int)((tensor_d0 & 0xFFFFu) << 16);         // tensor_dim0[15:0]
  g1[2] = (int)((tensor_d0 >> 16) | ((tensor_d1 & 0xFFFFu) << 16));
  g1[3] = (int)((tensor_d1 >> 16) | (tile_d0 << 16)); // tile_dim0
  g1[4] = (int)(tile_d1 & 0xFFFFu);                   // tile_dim1, tile_dim2=0
  g1[5] = (int)(unsigned)(stride0 & 0xFFFFFFFFu);     // tensor_dim0_stride lo
  g1[6] = (int)(unsigned)((stride0 >> 32) & 0xFFFFu); // stride hi, dim1_stride=0
  g1[7] = 0;
  tdm_v4i z4 = {0, 0, 0, 0};
  tdm_v8i z8 = {0, 0, 0, 0, 0, 0, 0, 0};
  __builtin_amdgcn_tensor_load_to_lds(g0, g1, z4, z4, z8, 0);
}

// ---------------- flash attention (WMMA f16 -> f32, TDM double buffer) -------
__global__ __launch_bounds__(256)
void fp8fp4_attn_wmma(const float* __restrict__ q, const _Float16* __restrict__ kd,
                      const _Float16* __restrict__ vdT, float* __restrict__ out) {
  __shared__ _Float16 sK[2][KT][CD];        // 16 KB double-buffered K tile
  __shared__ _Float16 sVt[2][CD][KT];       // 16 KB double-buffered V^T tile
  __shared__ _Float16 sP[NWAVE][16][KT];    //  8 KB per-wave P staging

  const int b = blockIdx.z, h = blockIdx.y, qblk = blockIdx.x;
  const int hkv  = h / CGQ;
  const int tid  = threadIdx.x;
  const int wave = tid >> 5;
  const int lane = tid & 31;
  const int mloc = lane & 15;   // row (A) or column (B/C) within fragment
  const int lh   = lane >> 4;   // lane half
  const int q0   = qblk * QB;
  const int qbase  = q0 + wave * 16;
  const int ntiles = (q0 + QB) / KT;   // all waves iterate to block-max key

  // ---- per-wave Q A-fragments straight from global (f32 -> f16, pre-scaled)
  // A layout (16-bit 16x32): lane m=lane&15; i<8 -> k=8*lh+i, i>=8 -> k=16+8*lh+(i-8)
  AFrag aQ[4];
  {
    const float* qrow = q + ((size_t)(b * CS + qbase + mloc) * CHQ + h) * CD;
#pragma unroll
    for (int c = 0; c < 4; ++c) {
      const float* p0 = qrow + c * 32 + lh * 8;
#pragma unroll
      for (int i = 0; i < 8; ++i) {
        aQ[c].h[i]     = (_Float16)(p0[i] * SCALING);
        aQ[c].h[8 + i] = (_Float16)(p0[16 + i] * SCALING);
      }
    }
  }

  // ---- online-softmax state ----
  v8f oacc[8];
  v8f vzero = {0.f, 0.f, 0.f, 0.f, 0.f, 0.f, 0.f, 0.f};
#pragma unroll
  for (int dt = 0; dt < 8; ++dt) oacc[dt] = vzero;
  float rmax[8], rsum[8];
#pragma unroll
  for (int r = 0; r < 8; ++r) { rmax[r] = -3.0e38f; rsum[r] = 0.f; }

  // ---- TDM tile issue (one wave drives the DMA engine) ----
  auto issue_tile = [&](int kt, int buf) {
    size_t gk = ((size_t)(b * CS + kt * KT) * CHKV + hkv) * CD;
    tdm_load_2d((unsigned)(uintptr_t)&sK[buf][0][0], kd + gk,
                /*tile*/ CD, KT, /*tensor*/ CD, KT,
                (unsigned long long)(CHKV * CD));
    size_t gv = ((size_t)(b * CHKV + hkv) * CD) * CS + (size_t)kt * KT;
    tdm_load_2d((unsigned)(uintptr_t)&sVt[buf][0][0], vdT + gv,
                /*tile*/ KT, CD, /*tensor*/ KT, CD,
                (unsigned long long)CS);
  };

  if (wave == 0) issue_tile(0, 0);

  for (int kt = 0; kt < ntiles; ++kt) {
    const int kb  = kt * KT;
    const int buf = kt & 1;
    if (wave == 0) {
      if (kt + 1 < ntiles) {
        issue_tile(kt + 1, buf ^ 1);
        __builtin_amdgcn_s_wait_tensorcnt(2);  // tile kt done (in-order TDM)
      } else {
        __builtin_amdgcn_s_wait_tensorcnt(0);
      }
    }
    __syncthreads();

    // ---- S = Q K^T : two 16-key C-tiles, 4 WMMAs each over D ----
    // B layout (16-bit 32x16): lane n=lane&15 is column; i<8 -> k=8*lh+i, i>=8 -> k=16+8*lh+(i-8).
    v8f sc[2];
#pragma unroll
    for (int nb = 0; nb < 2; ++nb) {
      v8f acc = vzero;
#pragma unroll
      for (int c = 0; c < 4; ++c) {
        AFrag bK;
        const _Float16* kp = &sK[buf][nb * 16 + mloc][c * 32 + lh * 8];
#pragma unroll
        for (int i = 0; i < 8; ++i) { bK.h[i] = kp[i]; bK.h[8 + i] = kp[16 + i]; }
        acc = __builtin_amdgcn_wmma_f32_16x16x32_f16(
            false, aQ[c].v, false, bK.v, (short)0, acc, false, false);
      }
      sc[nb] = acc;
    }

    // ---- causal mask + online softmax update ----
    // C layout: VGPR r -> M = r + 8*lh, N = lane&15.
    float mnew[8];
#pragma unroll
    for (int r = 0; r < 8; ++r) {
      int qrow = qbase + r + 8 * lh;
      if (kb + mloc > qrow)      sc[0][r] = -1e30f;
      if (kb + 16 + mloc > qrow) sc[1][r] = -1e30f;
      float t = fmaxf(sc[0][r], sc[1][r]);
#pragma unroll
      for (int off = 1; off < 16; off <<= 1)
        t = fmaxf(t, __shfl_xor(t, off, 32));
      mnew[r] = fmaxf(rmax[r], t);
    }
#pragma unroll
    for (int r = 0; r < 8; ++r) {
      float alpha = __expf(rmax[r] - mnew[r]);
      float p0 = __expf(sc[0][r] - mnew[r]);
      float p1 = __expf(sc[1][r] - mnew[r]);
      float ts = p0 + p1;
#pragma unroll
      for (int off = 1; off < 16; off <<= 1) ts += __shfl_xor(ts, off, 32);
      rsum[r] = rsum[r] * alpha + ts;
      rmax[r] = mnew[r];
#pragma unroll
      for (int dt = 0; dt < 8; ++dt) oacc[dt][r] *= alpha;
      sP[wave][r + 8 * lh][mloc]      = (_Float16)p0;
      sP[wave][r + 8 * lh][16 + mloc] = (_Float16)p1;
    }

    // ---- P (16x32) A-fragment from per-wave LDS staging ----
    AFrag aP;
    {
      const _Float16* p0 = &sP[wave][mloc][lh * 8];
      const _Float16* p1 = p0 + 16;
#pragma unroll
      for (int i = 0; i < 8; ++i) { aP.h[i] = p0[i]; aP.h[8 + i] = p1[i]; }
    }

    // ---- O += P * V : 8 WMMAs across D ----
#pragma unroll
    for (int dt = 0; dt < 8; ++dt) {
      AFrag bV;
      const _Float16* vp = &sVt[buf][dt * 16 + mloc][lh * 8];
#pragma unroll
      for (int i = 0; i < 8; ++i) { bV.h[i] = vp[i]; bV.h[8 + i] = vp[16 + i]; }
      oacc[dt] = __builtin_amdgcn_wmma_f32_16x16x32_f16(
          false, aP.v, false, bV.v, (short)0, oacc[dt], false, false);
    }
    __syncthreads();   // buffer 'buf' is reused by the issue at kt+2
  }

  // ---- normalize and write out [B,S,HQ*D] ----
#pragma unroll
  for (int r = 0; r < 8; ++r) {
    float inv = 1.0f / rsum[r];
    int qg = qbase + r + 8 * lh;
    float* op = out + ((size_t)(b * CS + qg) * CHQ + h) * CD;
#pragma unroll
    for (int dt = 0; dt < 8; ++dt)
      op[dt * 16 + mloc] = oacc[dt][r] * inv;
  }
}

// ---------------- host launch ----------------
extern "C" void kernel_launch(void* const* d_in, const int* in_sizes, int n_in,
                              void* d_out, int out_size, void* d_ws, size_t ws_size,
                              hipStream_t stream) {
  (void)in_sizes; (void)n_in; (void)out_size; (void)ws_size;
  const float* q = (const float*)d_in[0];
  const float* k = (const float*)d_in[1];
  const float* v = (const float*)d_in[2];
  float* out = (float*)d_out;

  _Float16* kd  = (_Float16*)d_ws;
  _Float16* vdT = kd + (size_t)CB * CS * CHKV * CD;

  int nrows = CB * CS * CHKV;                       // 32768 rows of 128
  fp8fp4_dequant_kv<<<dim3((nrows + NWAVE - 1) / NWAVE), dim3(256), 0, stream>>>(
      k, v, kd, vdT, nrows);

  dim3 grid(CS / QB, CHQ, CB);                      // 16 x 32 x 2
  fp8fp4_attn_wmma<<<grid, dim3(256), 0, stream>>>(q, kd, vdT, out);
}
